// MultiheadAttention_56272661512695
// MI455X (gfx1250) — matrix operand order repaired and checked by reading, not compiled, not run on hardware
//
#include <hip/hip_runtime.h>

typedef __attribute__((ext_vector_type(16))) _Float16 v16h;
typedef __attribute__((ext_vector_type(8)))  _Float16 v8h;
typedef __attribute__((ext_vector_type(4)))  _Float16 v4h;
typedef __attribute__((ext_vector_type(8)))  float    v8f;
typedef __attribute__((ext_vector_type(4)))  float    v4f;
typedef __attribute__((ext_vector_type(4)))  int      v4i;

#define TSZ 1024
#define BSZ 8
#define EMB 1024
#define NH  16
#define HD  64
#define NROW (TSZ*BSZ)   // 8192

static __device__ __forceinline__ v8f wmma16(v16h a, v16h b, v8f c) {
  // (neg_a, A, neg_b, B, c_mod, C, reuse_a, reuse_b)
  return __builtin_amdgcn_wmma_f32_16x16x32_f16(false, a, false, b, (short)0, c, false, false);
}

// ---------------------------------------------------------------------------
// Async global -> LDS copy (16B per lane), ASYNCcnt-tracked.  Builtin expects
// (AS1 int4*, AS3 int4*, imm offset, imm cpol) per the clang diagnostic.
// Falls back to a synchronous VGPR round-trip copy if the builtin is absent.
// ---------------------------------------------------------------------------
static __device__ __forceinline__ void async_copy_b128(const _Float16* g, _Float16* l)
{
#if __has_builtin(__builtin_amdgcn_global_load_async_to_lds_b128)
  __builtin_amdgcn_global_load_async_to_lds_b128(
      (__attribute__((address_space(1))) v4i*)(size_t)(const void*)g,
      (__attribute__((address_space(3))) v4i*)(unsigned)(size_t)(void*)l,
      0, 0);
#else
  *(v8h*)l = *(const v8h*)g;
#endif
}

template <int N>
static __device__ __forceinline__ void wait_asynccnt()
{
#if __has_builtin(__builtin_amdgcn_s_wait_asynccnt)
  __builtin_amdgcn_s_wait_asynccnt(N);
#else
  asm volatile("s_wait_asynccnt %0" :: "i"(N) : "memory");
#endif
}

// ---------------------------------------------------------------------------
// Kernel 0: fp32 -> f16 weight conversion (run once per launch, tiny)
// ---------------------------------------------------------------------------
__global__ __launch_bounds__(256)
void mha_cvt_f16_kernel(const float* __restrict__ src, _Float16* __restrict__ dst, int n)
{
  int i = (blockIdx.x * 256 + threadIdx.x) * 4;
  if (i + 3 < n) {
    v4f x = *(const v4f*)(src + i);
    v4h h;
    h[0] = (_Float16)x[0]; h[1] = (_Float16)x[1];
    h[2] = (_Float16)x[2]; h[3] = (_Float16)x[3];
    *(v4h*)(dst + i) = h;
  }
}

// ---------------------------------------------------------------------------
// Kernel 1: q/k/v in-projection GEMM (M=8192, N=3*1024, K=1024), f16 WMMA.
//   grid = (64, 48), block = 256 (8 waves).  Each wave: 16 rows x 64 cols.
//   Q gets scale 0.125 * log2(e) folded in (for exp2-based softmax).
//   Q,K stored head-major [bh][t][d]; V stored transposed [bh][d][t].
// ---------------------------------------------------------------------------
__global__ __launch_bounds__(256)
void mha_proj_kernel(const float* __restrict__ qin, const float* __restrict__ kin,
                     const float* __restrict__ vin, const _Float16* __restrict__ Wf,
                     const float* __restrict__ bias,
                     _Float16* __restrict__ Qh, _Float16* __restrict__ Kh,
                     _Float16* __restrict__ Vt)
{
  const int lane = threadIdx.x & 31;
  const int wave = threadIdx.x >> 5;
  const int hl   = lane >> 4;     // wave half (0/1)
  const int l16  = lane & 15;

  const int rowBase = blockIdx.x * 128 + wave * 16;
  const int proj    = blockIdx.y >> 4;            // 0=q 1=k 2=v
  const int nBase   = (blockIdx.y & 15) * 64;     // col tile inside projection

  const float* src  = (proj == 0) ? qin : (proj == 1) ? kin : vin;
  const float* arow = src + (size_t)(rowBase + l16) * EMB;

  v8f acc[4] = {};

  for (int kb = 0; kb < EMB; kb += 32) {
    // A fragment: lane m = l16; elems 0-15 <- K=kb+16*hl..+15
    const float* p0 = arow + kb + 16 * hl;
    v4f a0 = *(const v4f*)(p0);
    v4f a1 = *(const v4f*)(p0 + 4);
    v4f a2 = *(const v4f*)(p0 + 8);
    v4f a3 = *(const v4f*)(p0 + 12);
    v16h a;
#pragma unroll
    for (int j = 0; j < 4; ++j) {
      a[j]      = (_Float16)a0[j];
      a[4 + j]  = (_Float16)a1[j];
      a[8 + j]  = (_Float16)a2[j];
      a[12 + j] = (_Float16)a3[j];
    }
    // B fragments: B[k][n] = W[n][k]; lane n holds 16 consecutive k (32B load)
#pragma unroll
    for (int f = 0; f < 4; ++f) {
      const _Float16* wr = Wf + (size_t)(proj * EMB + nBase + f * 16 + l16) * EMB
                              + kb + 16 * hl;
      v16h b = *(const v16h*)wr;
      acc[f] = wmma16(a, b, acc[f]);
    }
  }

  const float scale = (proj == 0) ? (0.125f * 1.44269504088896f) : 1.0f;
#pragma unroll
  for (int f = 0; f < 4; ++f) {
    const int n  = nBase + f * 16 + l16;
    const float bv = bias[proj * EMB + n];
    const int h = n >> 6, d = n & 63;
#pragma unroll
    for (int i = 0; i < 8; ++i) {
      const int r = rowBase + i + 8 * hl;       // row = t*BSZ + b
      const int t = r >> 3, b = r & 7;
      const int bh = b * NH + h;
      const _Float16 val = (_Float16)((acc[f][i] + bv) * scale);
      if (proj == 2)       Vt[((size_t)bh * HD + d) * TSZ + t] = val;
      else if (proj == 1)  Kh[((size_t)bh * TSZ + t) * HD + d] = val;
      else                 Qh[((size_t)bh * TSZ + t) * HD + d] = val;
    }
  }
}

// ---------------------------------------------------------------------------
// Kernel 2: flash attention with async-LDS staged K/V, double buffered.
//   grid = (16, 128 heads), block = 128 (4 waves).  Block covers 64 queries
//   of one head; all 4 waves share the K/V tiles staged in LDS (4x traffic
//   reduction vs per-wave global loads).  64-key steps; online softmax in
//   base-2 domain (scale folded into Q projection).
// ---------------------------------------------------------------------------
static __device__ __forceinline__ void stage_tiles(const _Float16* __restrict__ Kh,
                                                   const _Float16* __restrict__ Vt,
                                                   _Float16* kl, _Float16* vl,
                                                   int bh, int kv, int tid)
{
  // K tile: 64 keys x 64 d  (row-major [key][d]) -- 8KB, 16B per lane per op
#pragma unroll
  for (int it = 0; it < 4; ++it) {
    const int c = it * 128 + tid;
    const int row = c >> 3, col = (c & 7) * 8;
    async_copy_b128(Kh + ((size_t)bh * TSZ + kv + row) * HD + col, kl + row * 64 + col);
  }
  // V tile: 64 d x 64 keys  (row-major [d][key]) from per-head transposed V
#pragma unroll
  for (int it = 0; it < 4; ++it) {
    const int c = it * 128 + tid;
    const int d = c >> 3, col = (c & 7) * 8;
    async_copy_b128(Vt + ((size_t)bh * HD + d) * TSZ + kv + col, vl + d * 64 + col);
  }
}

__global__ __launch_bounds__(128)
void mha_attn_kernel(const _Float16* __restrict__ Qh, const _Float16* __restrict__ Kh,
                     const _Float16* __restrict__ Vt, _Float16* __restrict__ Af)
{
  __shared__ _Float16 KL[2][64 * 64];   // 2 x 8KB, [key][d]
  __shared__ _Float16 VL[2][64 * 64];   // 2 x 8KB, [d][key]
  __shared__ _Float16 PL[4][16 * 64];   // per-wave P-tile transpose buffer

  const int tid  = threadIdx.x;
  const int lane = tid & 31;
  const int wave = tid >> 5;
  const int hl   = lane >> 4;
  const int l16  = lane & 15;

  const int bh    = blockIdx.y;                   // b*NH + h
  const int qBase = (blockIdx.x * 4 + wave) * 16;

  // Persistent Q A-fragments (K = head dim 64, split 2 x 32)
  const _Float16* qrow = Qh + ((size_t)bh * TSZ + qBase + l16) * HD;
  v16h qa[2];
#pragma unroll
  for (int ks = 0; ks < 2; ++ks) {
    v8h lo = *(const v8h*)(qrow + ks * 32 + 16 * hl);
    v8h hi = *(const v8h*)(qrow + ks * 32 + 16 * hl + 8);
#pragma unroll
    for (int j = 0; j < 8; ++j) { qa[ks][j] = lo[j]; qa[ks][8 + j] = hi[j]; }
  }

  v8f oacc[4] = {};
  float mrun[8], lrun[8];
#pragma unroll
  for (int i = 0; i < 8; ++i) { mrun[i] = -3.0e38f; lrun[i] = 0.0f; }

  _Float16* pl = &PL[wave][0];

  constexpr int NSTEP = TSZ / 64;   // 16
  // Prefetch pipeline, distance 2 (8 async ops per wave per step)
  stage_tiles(Kh, Vt, &KL[0][0], &VL[0][0], bh, 0, tid);
  stage_tiles(Kh, Vt, &KL[1][0], &VL[1][0], bh, 64, tid);

  for (int s = 0; s < NSTEP; ++s) {
    const int buf = s & 1;
    // Async loads complete in order: <=8 outstanding => step-s tiles landed.
    if (s < NSTEP - 1) wait_asynccnt<8>(); else wait_asynccnt<0>();
    __syncthreads();                 // all waves' tiles visible

    const _Float16* kl = &KL[buf][0];
    const _Float16* vl = &VL[buf][0];

    // ---- scores: S[16x64] = Q . K^T, 4 groups of 16 keys, B from LDS ----
    v8f sg[4];
#pragma unroll
    for (int g = 0; g < 4; ++g) {
      v8f sc = {};
      const _Float16* krow = kl + (g * 16 + l16) * 64;
#pragma unroll
      for (int ks = 0; ks < 2; ++ks) {
        v16h kb = *(const v16h*)(krow + ks * 32 + 16 * hl);
        sc = wmma16(qa[ks], kb, sc);
      }
      sg[g] = sc;
    }

    // ---- row max over 64-key tile (reduce across the 16 lanes of half) ----
    float rmx[8];
#pragma unroll
    for (int i = 0; i < 8; ++i)
      rmx[i] = fmaxf(fmaxf(sg[0][i], sg[1][i]), fmaxf(sg[2][i], sg[3][i]));
#pragma unroll
    for (int off = 8; off >= 1; off >>= 1) {
#pragma unroll
      for (int i = 0; i < 8; ++i)
        rmx[i] = fmaxf(rmx[i], __shfl_xor(rmx[i], off, 32));
    }

    // ---- online softmax update (base-2 domain) ----
    float pv[4][8], rs[8];
#pragma unroll
    for (int i = 0; i < 8; ++i) {
      const float mnew = fmaxf(mrun[i], rmx[i]);
      const float corr = exp2f(mrun[i] - mnew);
      mrun[i] = mnew;
      lrun[i] *= corr;
#pragma unroll
      for (int f = 0; f < 4; ++f) oacc[f][i] *= corr;
      rs[i] = 0.0f;
#pragma unroll
      for (int g = 0; g < 4; ++g) { pv[g][i] = exp2f(sg[g][i] - mnew); rs[i] += pv[g][i]; }
    }
#pragma unroll
    for (int off = 8; off >= 1; off >>= 1) {
#pragma unroll
      for (int i = 0; i < 8; ++i)
        rs[i] += __shfl_xor(rs[i], off, 32);
    }
#pragma unroll
    for (int i = 0; i < 8; ++i) lrun[i] += rs[i];

    // ---- transpose P (C layout -> A layout) via per-wave LDS ----
#pragma unroll
    for (int g = 0; g < 4; ++g) {
#pragma unroll
      for (int i = 0; i < 8; ++i)
        pl[(i + 8 * hl) * 64 + g * 16 + l16] = (_Float16)pv[g][i];
    }
    v16h pa[2];
#pragma unroll
    for (int ss = 0; ss < 2; ++ss) {
      v8h lo = *(const v8h*)(pl + l16 * 64 + ss * 32 + 16 * hl);
      v8h hi = *(const v8h*)(pl + l16 * 64 + ss * 32 + 16 * hl + 8);
#pragma unroll
      for (int j = 0; j < 8; ++j) { pa[ss][j] = lo[j]; pa[ss][8 + j] = hi[j]; }
    }

    // ---- O[16x64] += P[16x64] . V[64x64], B from LDS V tile ----
#pragma unroll
    for (int ss = 0; ss < 2; ++ss) {
#pragma unroll
      for (int f = 0; f < 4; ++f) {
        v16h vb = *(const v16h*)(vl + (f * 16 + l16) * 64 + ss * 32 + 16 * hl);
        oacc[f] = wmma16(pa[ss], vb, oacc[f]);
      }
    }

    __syncthreads();                 // everyone done reading buf before reuse
    if (s + 2 < NSTEP)
      stage_tiles(Kh, Vt, &KL[buf][0], &VL[buf][0], bh, (s + 2) * 64, tid);
  }

  // ---- normalize, store to (t, b, e) f16 layout for out-projection ----
  const int b = bh >> 4, h = bh & 15;
#pragma unroll
  for (int f = 0; f < 4; ++f) {
    const int e = h * HD + f * 16 + l16;
#pragma unroll
    for (int i = 0; i < 8; ++i) {
      const int t = qBase + i + 8 * hl;
      Af[(size_t)(t * BSZ + b) * EMB + e] = (_Float16)(oacc[f][i] / lrun[i]);
    }
  }
}

// ---------------------------------------------------------------------------
// Kernel 3: out projection (M=8192, N=1024, K=1024), bias add, fp32 out.
//   grid = (64, 16), block = 256 (8 waves), wave tile 16x64.
// ---------------------------------------------------------------------------
__global__ __launch_bounds__(256)
void mha_outproj_kernel(const _Float16* __restrict__ Af, const _Float16* __restrict__ Wf,
                        const float* __restrict__ bias, float* __restrict__ out)
{
  const int lane = threadIdx.x & 31;
  const int wave = threadIdx.x >> 5;
  const int hl   = lane >> 4;
  const int l16  = lane & 15;

  const int rowBase = blockIdx.x * 128 + wave * 16;
  const int nBase   = blockIdx.y * 64;

  const _Float16* arow = Af + (size_t)(rowBase + l16) * EMB;
  v8f acc[4] = {};

  for (int kb = 0; kb < EMB; kb += 32) {
    v8h lo = *(const v8h*)(arow + kb + 16 * hl);
    v8h hi = *(const v8h*)(arow + kb + 16 * hl + 8);
    v16h a;
#pragma unroll
    for (int j = 0; j < 8; ++j) { a[j] = lo[j]; a[8 + j] = hi[j]; }
#pragma unroll
    for (int f = 0; f < 4; ++f) {
      const _Float16* wr = Wf + (size_t)(nBase + f * 16 + l16) * EMB + kb + 16 * hl;
      v16h b = *(const v16h*)wr;
      acc[f] = wmma16(a, b, acc[f]);
    }
  }

#pragma unroll
  for (int f = 0; f < 4; ++f) {
    const int n = nBase + f * 16 + l16;
    const float bv = bias[f ? n : n];  // keep simple; bias indexed by n
    (void)bv;
    const float bvv = bias[n];
#pragma unroll
    for (int i = 0; i < 8; ++i) {
      const int r = rowBase + i + 8 * hl;
      out[(size_t)r * EMB + n] = acc[f][i] + bvv;
    }
  }
}

// ---------------------------------------------------------------------------
// Launcher.  Workspace (f16 elems): Qh|Kh|Vt|Af 4x8,388,608; Wif 3M; Wof 1M.
// ---------------------------------------------------------------------------
extern "C" void kernel_launch(void* const* d_in, const int* in_sizes, int n_in,
                              void* d_out, int out_size, void* d_ws, size_t ws_size,
                              hipStream_t stream)
{
  (void)in_sizes; (void)n_in; (void)out_size; (void)ws_size;

  const float* q  = (const float*)d_in[0];
  const float* k  = (const float*)d_in[1];
  const float* v  = (const float*)d_in[2];
  const float* Wi = (const float*)d_in[3];
  const float* bi = (const float*)d_in[4];
  const float* Wo = (const float*)d_in[5];
  const float* bo = (const float*)d_in[6];
  float* out = (float*)d_out;

  const size_t HSZ = (size_t)BSZ * NH * TSZ * HD;   // 8,388,608 halfs
  _Float16* Qh  = (_Float16*)d_ws;
  _Float16* Kh  = Qh + HSZ;
  _Float16* Vt  = Kh + HSZ;
  _Float16* Af  = Vt + HSZ;
  _Float16* Wif = Af + HSZ;
  _Float16* Wof = Wif + (size_t)3 * EMB * EMB;

  mha_cvt_f16_kernel<<<dim3((3 * EMB * EMB) / 1024), 256, 0, stream>>>(Wi, Wif, 3 * EMB * EMB);
  mha_cvt_f16_kernel<<<dim3((EMB * EMB) / 1024),     256, 0, stream>>>(Wo, Wof, EMB * EMB);

  mha_proj_kernel<<<dim3(NROW / 128, 48), 256, 0, stream>>>(q, k, v, Wif, bi, Qh, Kh, Vt);
  mha_attn_kernel<<<dim3(TSZ / 64, BSZ * NH), 128, 0, stream>>>(Qh, Kh, Vt, Af);
  mha_outproj_kernel<<<dim3(NROW / 128, EMB / 64), 256, 0, stream>>>(Af, Wof, bo, out);
}
